// RNNARModel_85109071938310
// MI455X (gfx1250) — compile-verified, matrix operand order checked by reading, not run on hardware
//
#include <hip/hip_runtime.h>
#include <hip/hip_bf16.h>

// ---------------- problem constants (from reference) ----------------
#define B_TOT    1024
#define L_ENC    2048
#define DEC_LEN  64
#define H        128
#define G4       512            // 4*H gate width
#define EMB      16
#define CARD     200
#define IN_RAW   18             // EMB + cont + x
#define IN_PAD   32             // padded to WMMA K granularity
#define KTOT     160            // IN_PAD + H
#define BT       16             // batch rows per block (WMMA M)
#define NTHREADS 256            // 8 wave32
#define XH_ELEMS (BT * KTOT)    // one xh buffer (bf16 elems)

typedef __attribute__((ext_vector_type(16))) __bf16 v16bf;
typedef __attribute__((ext_vector_type(8)))  float  v8f;

union FragU { v16bf v; uint4 q[2]; };

__device__ __forceinline__ unsigned short f2bf(float f) {
    unsigned int u = __float_as_uint(f);
    u += 0x7FFFu + ((u >> 16) & 1u);          // round-to-nearest-even
    return (unsigned short)(u >> 16);
}
// hardware transcendentals only: v_exp_f32 / v_rcp_f32 (no IEEE div sequences)
__device__ __forceinline__ float frcp(float x)  { return __builtin_amdgcn_rcpf(x); }
__device__ __forceinline__ float sigm(float x)  { return frcp(1.f + __expf(-x)); }
__device__ __forceinline__ float ftanh(float x) {
    float e = __expf(2.f * x);
    return (e - 1.f) * frcp(e + 1.f);
}

// ---------------- LDS layout (bytes) ----------------
#define OFF_W     0                                   // bf16 [G4][KTOT]   (staging for reg weights)
#define OFF_XH    (OFF_W    + G4 * KTOT * 2)          // bf16 2 x [BT][KTOT] double buffer
#define OFF_EMB   (OFF_XH   + 2 * XH_ELEMS * 2)       // f32  [CARD][EMB]
#define OFF_BIAS  (OFF_EMB  + CARD * EMB * 4)         // f32  [G4]
#define OFF_HF    (OFF_BIAS + G4 * 4)                 // f32  [BT][H]  (decoder heads)
#define OFF_HW    (OFF_HF   + BT * H * 4)             // f32  [6][H]
#define OFF_HB    (OFF_HW   + 6 * H * 4)              // f32  [8]
#define SMEM_BYTES (OFF_HB  + 8 * 4)                  // ~196 KB < 320 KB/WGP

__device__ __forceinline__ void load_weights(unsigned short* __restrict__ wLds,
                                             float* __restrict__ biasLds,
                                             const float* __restrict__ Wih,
                                             const float* __restrict__ Whh,
                                             const float* __restrict__ b, int tid) {
    for (int idx = tid; idx < G4 * KTOT; idx += NTHREADS) {
        int n = idx / KTOT, k = idx % KTOT;
        float w = 0.f;
        if (k < IN_RAW)       w = Wih[n * IN_RAW + k];
        else if (k >= IN_PAD) w = Whh[n * H + (k - IN_PAD)];
        wLds[idx] = f2bf(w);
    }
    for (int n = tid; n < G4; n += NTHREADS) biasLds[n] = b[n];
}

// ---------------- fused encoder + decoder, one block per 16-row batch tile ----------------
extern "C" __global__ void __launch_bounds__(NTHREADS)
rnnar_fused(const int* __restrict__ cat_in, const float* __restrict__ cont_in,
            const float* __restrict__ X_in, const int* __restrict__ cat_out,
            const float* __restrict__ cont_out, const float* __restrict__ emb_table,
            const float* __restrict__ cont_w,
            const float* __restrict__ Wih_e, const float* __restrict__ Whh_e, const float* __restrict__ b_e,
            const float* __restrict__ Wih_d, const float* __restrict__ Whh_d, const float* __restrict__ b_d,
            const float* __restrict__ Wm, const float* __restrict__ bm,
            const float* __restrict__ Ws, const float* __restrict__ bs,
            const float* __restrict__ Wv, const float* __restrict__ bv,
            float* __restrict__ out)
{
    extern __shared__ char smem[];
    unsigned short* wLds   = (unsigned short*)(smem + OFF_W);
    unsigned short* xhAll  = (unsigned short*)(smem + OFF_XH);   // two buffers
    float* embLds  = (float*)(smem + OFF_EMB);
    float* biasLds = (float*)(smem + OFF_BIAS);
    float* hLds    = (float*)(smem + OFF_HF);
    float* headW   = (float*)(smem + OFF_HW);
    float* headB   = (float*)(smem + OFF_HB);

    const int tid  = threadIdx.x;
    const int wv   = tid >> 5;
    const int lane = tid & 31;
    const int m    = lane & 15;
    const int half = lane >> 4;
    const int kb   = half * 8;          // ISA 16-bit fragment K split per lane-half
    const int j    = wv * 16 + m;       // hidden column this lane owns (all 4 gates)
    const int row0 = blockIdx.x * BT;
    const float w00 = cont_w[0];

    // per-thread input-fill jobs (BT*IN_RAW = 288 = 256 + 32)
    const int  r1 = tid / IN_RAW, c1 = tid % IN_RAW;
    const bool has2 = tid < (BT * IN_RAW - NTHREADS);
    const int  r2 = (tid + NTHREADS) / IN_RAW, c2 = (tid + NTHREADS) % IN_RAW;

    // ---- resident tables, encoder weights (bf16) staged in LDS ----
    for (int i = tid; i < CARD * EMB; i += NTHREADS) embLds[i] = emb_table[i];
    load_weights(wLds, biasLds, Wih_e, Whh_e, b_e, tid);
    for (int u = tid; u < 2 * XH_ELEMS; u += NTHREADS) xhAll[u] = 0;  // zeros incl. pad cols
    __syncthreads();

    // ---- loop-invariant state in registers: B fragments, biases, cell state ----
    FragU bw[4][5];
#pragma unroll
    for (int g = 0; g < 4; ++g)
#pragma unroll
        for (int kc = 0; kc < 5; ++kc) {
            const unsigned short* p = wLds + (g * H + j) * KTOT + kc * 32 + kb;
            bw[g][kc].q[0] = *(const uint4*)(p);
            bw[g][kc].q[1] = *(const uint4*)(p + 16);
        }
    float bias_r[4];
#pragma unroll
    for (int g = 0; g < 4; ++g) bias_r[g] = biasLds[g * H + j];
    float creg[8];
#pragma unroll
    for (int r = 0; r < 8; ++r) creg[r] = 0.f;

    // x-part of buffer 0 for t = 0
    {
        int g1 = (row0 + r1) * L_ENC;
        float v1 = (c1 < EMB) ? embLds[cat_in[g1] * EMB + c1]
                 : (c1 == EMB) ? cont_in[g1] * w00 : X_in[g1];
        xhAll[r1 * KTOT + c1] = f2bf(v1);
        if (has2) {
            int g2 = (row0 + r2) * L_ENC;
            float v2 = (c2 < EMB) ? embLds[cat_in[g2] * EMB + c2]
                     : (c2 == EMB) ? cont_in[g2] * w00 : X_in[g2];
            xhAll[r2 * KTOT + c2] = f2bf(v2);
        }
    }
    __syncthreads();

    // ================= encoder: 2048 sequential steps, ONE barrier each =================
    for (int t = 0; t < L_ENC; ++t) {
        const unsigned short* xin  = xhAll + (t & 1) * XH_ELEMS;
        unsigned short*       xout = xhAll + ((t + 1) & 1) * XH_ELEMS;

        // prefetch next-step inputs (HBM latency hides behind WMMA chain)
        const bool pref = (t + 1 < L_ENC);
        int pc1 = 0, pc2 = 0; float pv1 = 0.f, pv2 = 0.f;
        if (pref) {
            int g1 = (row0 + r1) * L_ENC + t + 1;
            if (c1 < EMB) pc1 = cat_in[g1]; else pv1 = (c1 == EMB) ? cont_in[g1] : X_in[g1];
            if (has2) {
                int g2 = (row0 + r2) * L_ENC + t + 1;
                if (c2 < EMB) pc2 = cat_in[g2]; else pv2 = (c2 == EMB) ? cont_in[g2] : X_in[g2];
            }
        }

        // A fragments for this step
        FragU a[5];
#pragma unroll
        for (int kc = 0; kc < 5; ++kc) {
            const unsigned short* p = xin + m * KTOT + kc * 32 + kb;
            a[kc].q[0] = *(const uint4*)(p);
            a[kc].q[1] = *(const uint4*)(p + 16);
        }

        // 4 gate tiles, same 16 hidden columns; bias pre-loaded into the accumulator
        v8f acc[4];
#pragma unroll
        for (int g = 0; g < 4; ++g) {
            const float bg = bias_r[g];
            v8f t0 = {bg, bg, bg, bg, bg, bg, bg, bg};
#pragma unroll
            for (int kc = 0; kc < 5; ++kc)
                t0 = __builtin_amdgcn_wmma_f32_16x16x32_bf16(
                         false, a[kc].v, false, bw[g][kc].v, (short)0, t0, false, false);
            acc[g] = t0;
        }

#pragma unroll
        for (int r = 0; r < 8; ++r) {
            float is = sigm(acc[0][r]);
            float fs = sigm(acc[1][r]);
            float gt = ftanh(acc[2][r]);
            float os = sigm(acc[3][r]);
            float cn = fs * creg[r] + is * gt;
            creg[r] = cn;
            float hn = os * ftanh(cn);
            xout[(half * 8 + r) * KTOT + IN_PAD + j] = f2bf(hn);   // bf16 h for next step
        }

        if (pref) {
            float v1 = (c1 < EMB) ? embLds[pc1 * EMB + c1] : (c1 == EMB) ? pv1 * w00 : pv1;
            xout[r1 * KTOT + c1] = f2bf(v1);
            if (has2) {
                float v2 = (c2 < EMB) ? embLds[pc2 * EMB + c2] : (c2 == EMB) ? pv2 * w00 : pv2;
                xout[r2 * KTOT + c2] = f2bf(v2);
            }
        }
        __syncthreads();
    }

    // ================= swap to decoder weights + heads =================
    load_weights(wLds, biasLds, Wih_d, Whh_d, b_d, tid);
    for (int i = tid; i < 6 * H; i += NTHREADS) {
        int o = i / H, k = i % H;
        headW[i] = (o == 0) ? Wm[k] : (o == 1) ? Ws[k] : Wv[(o - 2) * H + k];
    }
    if (tid < 6) headB[tid] = (tid == 0) ? bm[0] : (tid == 1) ? bs[0] : bv[tid - 2];
    __syncthreads();

#pragma unroll
    for (int g = 0; g < 4; ++g)
#pragma unroll
        for (int kc = 0; kc < 5; ++kc) {
            const unsigned short* p = wLds + (g * H + j) * KTOT + kc * 32 + kb;
            bw[g][kc].q[0] = *(const uint4*)(p);
            bw[g][kc].q[1] = *(const uint4*)(p + 16);
        }
#pragma unroll
    for (int g = 0; g < 4; ++g) bias_r[g] = biasLds[g * H + j];

    // decoder initial x-part into buffer 0 (h-part already there from encoder step 2047)
    {
        int g1 = (row0 + r1) * L_ENC + (L_ENC - 1);
        float v1 = (c1 < EMB) ? embLds[cat_in[g1] * EMB + c1]
                 : (c1 == EMB) ? cont_in[g1] * w00 : X_in[g1];
        xhAll[r1 * KTOT + c1] = f2bf(v1);
        if (has2) {
            int g2 = (row0 + r2) * L_ENC + (L_ENC - 1);
            float v2 = (c2 < EMB) ? embLds[cat_in[g2] * EMB + c2]
                     : (c2 == EMB) ? cont_in[g2] * w00 : X_in[g2];
            xhAll[r2 * KTOT + c2] = f2bf(v2);
        }
    }
    __syncthreads();

    // ================= decoder: 64 autoregressive steps, two barriers each =================
    for (int td = 0; td < DEC_LEN; ++td) {
        const unsigned short* xin  = xhAll + (td & 1) * XH_ELEMS;
        unsigned short*       xout = xhAll + ((td + 1) & 1) * XH_ELEMS;

        FragU a[5];
#pragma unroll
        for (int kc = 0; kc < 5; ++kc) {
            const unsigned short* p = xin + m * KTOT + kc * 32 + kb;
            a[kc].q[0] = *(const uint4*)(p);
            a[kc].q[1] = *(const uint4*)(p + 16);
        }

        v8f acc[4];
#pragma unroll
        for (int g = 0; g < 4; ++g) {
            const float bg = bias_r[g];
            v8f t0 = {bg, bg, bg, bg, bg, bg, bg, bg};
#pragma unroll
            for (int kc = 0; kc < 5; ++kc)
                t0 = __builtin_amdgcn_wmma_f32_16x16x32_bf16(
                         false, a[kc].v, false, bw[g][kc].v, (short)0, t0, false, false);
            acc[g] = t0;
        }

#pragma unroll
        for (int r = 0; r < 8; ++r) {
            float is = sigm(acc[0][r]);
            float fs = sigm(acc[1][r]);
            float gt = ftanh(acc[2][r]);
            float os = sigm(acc[3][r]);
            float cn = fs * creg[r] + is * gt;
            creg[r] = cn;
            float hn = os * ftanh(cn);
            int row = half * 8 + r;
            xout[row * KTOT + IN_PAD + j] = f2bf(hn);  // bf16 h for next matmul
            hLds[row * H + j] = hn;                    // f32 h for heads
        }
        __syncthreads();

        // heads: mu / softplus(std) / v[4]   (96 lanes of 128-dots)
        if (tid < BT * 6) {
            int r = tid / 6, o = tid % 6;
            const float* hr = hLds + r * H;
            const float* wp = headW + o * H;
            float acch = headB[o];
#pragma unroll 8
            for (int k = 0; k < H; ++k) acch += hr[k] * wp[k];
            int orow = (row0 + r) * DEC_LEN + td;
            if (o == 0) {
                out[orow] = acch;
                xout[r * KTOT + 17] = f2bf(acch);      // mu feeds next x_prev
            } else if (o == 1) {
                float sp = (acch > 0.f) ? acch + log1pf(__expf(-acch)) : log1pf(__expf(acch));
                out[B_TOT * DEC_LEN + orow] = sp;
            } else {
                out[2 * B_TOT * DEC_LEN + orow * 4 + (o - 2)] = acch;
            }
        }
        // decoder-known features at step td become cols 0..16 of next step's x
        for (int idx = tid; idx < BT * 17; idx += NTHREADS) {
            int r = idx / 17, c = idx % 17;
            int g = (row0 + r) * DEC_LEN + td;
            float v = (c < EMB) ? embLds[cat_out[g] * EMB + c] : cont_out[g] * w00;
            xout[r * KTOT + c] = f2bf(v);
        }
        __syncthreads();
    }
}

extern "C" void kernel_launch(void* const* d_in, const int* in_sizes, int n_in,
                              void* d_out, int out_size, void* d_ws, size_t ws_size,
                              hipStream_t stream) {
    (void)in_sizes; (void)n_in; (void)out_size; (void)d_ws; (void)ws_size;
    hipFuncSetAttribute(reinterpret_cast<const void*>(rnnar_fused),
                        hipFuncAttributeMaxDynamicSharedMemorySize, (int)SMEM_BYTES);
    rnnar_fused<<<dim3(B_TOT / BT), dim3(NTHREADS), SMEM_BYTES, stream>>>(
        (const int*)d_in[0],  (const float*)d_in[1],  (const float*)d_in[2],
        (const int*)d_in[3],  (const float*)d_in[4],  (const float*)d_in[5],
        (const float*)d_in[6],
        (const float*)d_in[7],  (const float*)d_in[8],  (const float*)d_in[9],
        (const float*)d_in[10], (const float*)d_in[11], (const float*)d_in[12],
        (const float*)d_in[13], (const float*)d_in[14], (const float*)d_in[15],
        (const float*)d_in[16], (const float*)d_in[17], (const float*)d_in[18],
        (float*)d_out);
}
